// Decoder_45475113730603
// MI455X (gfx1250) — compile-verified, matrix operand order checked by reading, not run on hardware
//
#include <hip/hip_runtime.h>

typedef __bf16 bf16_t;
typedef __bf16 v16bf __attribute__((ext_vector_type(16)));
typedef float  v8f   __attribute__((ext_vector_type(8)));
typedef unsigned int v4u __attribute__((ext_vector_type(4)));

#define STEPS 400
#define BT 16          // batch rows per workgroup
#define TENC 256
#define DDIM 256
#define NMEL 80
#define WG_THREADS 512 // 16 waves (wave32)

// ---------------------------------------------------------------------------
// Fragment loaders (wave32 WMMA 16x16x32 bf16 layouts)
// A 16x32: lane l -> row m=l&15, K elements {c*8+0..7, 16+c*8+0..7}, c=l>>4
// B 32x16: lane l -> col n=l&15, same K striping (weights pre-swizzled so each
//          lane reads 32 contiguous bytes)
// C/D 16x16 f32: lane l -> col n=l&15, VGPR r -> row m=r+8*(l>>4)
// ---------------------------------------------------------------------------

__device__ inline v16bf load_frag_lds(const bf16_t* base, int ldk, int kt, int lane) {
  int m = lane & 15;
  int koff = kt * 32 + ((lane >> 4) << 3);
  const v4u* p0 = (const v4u*)(base + m * ldk + koff);
  const v4u* p1 = (const v4u*)(base + m * ldk + koff + 16);
  union { v4u u[2]; v16bf v; } t;
  t.u[0] = *p0;
  t.u[1] = *p1;
  return t.v;
}

__device__ inline v16bf load_frag_w(const bf16_t* fr, int Ktiles, int nt, int kt, int lane) {
  const bf16_t* f = fr + ((((size_t)nt * Ktiles + kt) << 5) + (size_t)lane) * 16;
  union { v4u u[2]; v16bf v; } t;
  t.u[0] = ((const v4u*)f)[0];
  t.u[1] = ((const v4u*)f)[1];
  return t.v;
}

__device__ inline v8f wmma_bf16(v16bf a, v16bf b, v8f c) {
  return __builtin_amdgcn_wmma_f32_16x16x32_bf16(false, a, false, b, (short)0, c,
                                                 false, false);
}

__device__ inline v8f zero8() {
  v8f z;
#pragma unroll
  for (int i = 0; i < 8; ++i) z[i] = 0.f;
  return z;
}

__device__ inline float sigm(float x) { return 1.f / (1.f + __expf(-x)); }

__device__ inline float wred_add(float v) {
#pragma unroll
  for (int o = 16; o >= 1; o >>= 1) v += __shfl_xor(v, o, 32);
  return v;
}
__device__ inline float wred_max(float v) {
#pragma unroll
  for (int o = 16; o >= 1; o >>= 1) v = fmaxf(v, __shfl_xor(v, o, 32));
  return v;
}

// ---------------------------------------------------------------------------
// GEMM tiles: C(16xN) = A(16xK) @ W^T fragments, epilogues fused
// ---------------------------------------------------------------------------

__device__ inline void gemm_relu_bf16(const bf16_t* A, int ldk, int Ktiles,
                                      const bf16_t* Wf, int Ntiles,
                                      const float* bias, bf16_t* Cout, int ldc,
                                      int wave, int lane) {
  for (int nt = wave; nt < Ntiles; nt += 16) {
    v8f acc = zero8();
    for (int kt = 0; kt < Ktiles; ++kt) {
      v16bf a = load_frag_lds(A, ldk, kt, lane);
      v16bf b = load_frag_w(Wf, Ktiles, nt, kt, lane);
      acc = wmma_bf16(a, b, acc);
    }
    int m0 = (lane >> 4) << 3;
    int n = nt * 16 + (lane & 15);
    float bv = bias[n];
#pragma unroll
    for (int r = 0; r < 8; ++r) {
      float v = acc[r] + bv;
      v = v > 0.f ? v : 0.f;
      Cout[(m0 + r) * ldc + n] = (bf16_t)v;
    }
  }
}

__device__ inline void gemm_bias_f32(const bf16_t* A, int ldk, int Ktiles,
                                     const bf16_t* Wf, int Ntiles,
                                     const float* bias, float* Cout, int ldc,
                                     int wave, int lane) {
  for (int nt = wave; nt < Ntiles; nt += 16) {
    v8f acc = zero8();
    for (int kt = 0; kt < Ktiles; ++kt) {
      v16bf a = load_frag_lds(A, ldk, kt, lane);
      v16bf b = load_frag_w(Wf, Ktiles, nt, kt, lane);
      acc = wmma_bf16(a, b, acc);
    }
    int m0 = (lane >> 4) << 3;
    int n = nt * 16 + (lane & 15);
    float bv = bias[n];
#pragma unroll
    for (int r = 0; r < 8; ++r) Cout[(m0 + r) * ldc + n] = acc[r] + bv;
  }
}

// Fused GRU cell: wave `wave` owns output cols [wave*16, wave*16+16).
// Accumulates r/z gates across both GEMMs; keeps i_n and h_n separate for
// n = tanh(i_n + r*h_n); updates h in-place (f32 LDS).
__device__ inline void gru_fused(const bf16_t* Ax, int ldkx, int Ktx,
                                 const bf16_t* Wih, const bf16_t* Ah,
                                 const bf16_t* Whh, const float* bih,
                                 const float* bhh, float* hbuf, int wave,
                                 int lane) {
  int nt = wave;  // 16 waves <-> 16 col-tiles of D=256
  v8f ar = zero8(), az = zero8(), ain = zero8(), ahn = zero8();
  for (int kt = 0; kt < 8; ++kt) {  // h @ Whh.T, K = 256
    v16bf a = load_frag_lds(Ah, 256, kt, lane);
    v16bf br = load_frag_w(Whh, 8, nt, kt, lane);
    v16bf bz = load_frag_w(Whh, 8, 16 + nt, kt, lane);
    v16bf bn = load_frag_w(Whh, 8, 32 + nt, kt, lane);
    ar = wmma_bf16(a, br, ar);
    az = wmma_bf16(a, bz, az);
    ahn = wmma_bf16(a, bn, ahn);
  }
  for (int kt = 0; kt < Ktx; ++kt) {  // x @ Wih.T
    v16bf a = load_frag_lds(Ax, ldkx, kt, lane);
    v16bf br = load_frag_w(Wih, Ktx, nt, kt, lane);
    v16bf bz = load_frag_w(Wih, Ktx, 16 + nt, kt, lane);
    v16bf bn = load_frag_w(Wih, Ktx, 32 + nt, kt, lane);
    ar = wmma_bf16(a, br, ar);
    az = wmma_bf16(a, bz, az);
    ain = wmma_bf16(a, bn, ain);
  }
  int m0 = (lane >> 4) << 3;
  int d = nt * 16 + (lane & 15);
  float br_ = bih[d] + bhh[d];
  float bz_ = bih[256 + d] + bhh[256 + d];
  float bin_ = bih[512 + d];
  float bhn_ = bhh[512 + d];
#pragma unroll
  for (int r = 0; r < 8; ++r) {
    float rr = sigm(ar[r] + br_);
    float zz = sigm(az[r] + bz_);
    float nn = tanhf(ain[r] + bin_ + rr * (ahn[r] + bhn_));
    float h = hbuf[(m0 + r) * 256 + d];
    hbuf[(m0 + r) * 256 + d] = (1.f - zz) * nn + zz * h;
  }
}

// ---------------------------------------------------------------------------
// Precompute kernels
// ---------------------------------------------------------------------------

// Pack weight W (N,K row-major, used as x@W.T) into per-fragment bf16 layout:
// frag[(nt*Ktiles+kt)*32 + lane][j] ; zero-pads K (e.g. 80 -> 96).
__global__ void pack_weight(const float* __restrict__ src, bf16_t* __restrict__ dst,
                            int N, int K, int Ktiles) {
  int tid = blockIdx.x * blockDim.x + threadIdx.x;
  int total = (N / 16) * Ktiles * 32;
  if (tid >= total) return;
  int l = tid & 31;
  int frag = tid >> 5;          // == nt*Ktiles + kt
  int kt = frag % Ktiles;
  int n = (frag / Ktiles) * 16 + (l & 15);
  bf16_t* out = dst + ((size_t)frag * 32 + l) * 16;
#pragma unroll
  for (int j = 0; j < 16; ++j) {
    int k = kt * 32 + ((j >> 3) << 4) + ((l >> 4) << 3) + (j & 7);
    float v = (k < K) ? src[(size_t)n * K + k] : 0.f;
    out[j] = (bf16_t)v;
  }
}

// keys[b,t,:] = enc[b,t,:] @ attn_wk.T + attn_bk   (one block per (b,t))
__global__ void keys_kernel(const float* __restrict__ enc,
                            const float* __restrict__ wk,
                            const float* __restrict__ bk,
                            float* __restrict__ keys) {
  int bt = blockIdx.x;
  int d = threadIdx.x;
  __shared__ float srow[DDIM];
  srow[d] = enc[(size_t)bt * DDIM + d];
  __syncthreads();
  const float* w = wk + (size_t)d * DDIM;
  float s = bk[d];
  for (int k = 0; k < DDIM; ++k) s += srow[k] * w[k];
  keys[(size_t)bt * DDIM + d] = s;
}

// ---------------------------------------------------------------------------
// Persistent decoder: 4 workgroups x 16 batch rows, 400 serial steps
// ---------------------------------------------------------------------------
__global__ __launch_bounds__(WG_THREADS, 1) void decoder_kernel(
    const float* __restrict__ enc, const float* __restrict__ dec,
    const float* __restrict__ keys, const float* __restrict__ pre_b1,
    const float* __restrict__ pre_b2, const float* __restrict__ attn_bq,
    const float* __restrict__ attn_v, const float* __restrict__ arnn_bih,
    const float* __restrict__ arnn_bhh, const float* __restrict__ g1_bih,
    const float* __restrict__ g1_bhh, const float* __restrict__ g2_bih,
    const float* __restrict__ g2_bhh, const float* __restrict__ fc_b,
    const bf16_t* __restrict__ Wp1, const bf16_t* __restrict__ Wp2,
    const bf16_t* __restrict__ Wai, const bf16_t* __restrict__ Wah,
    const bf16_t* __restrict__ Wq, const bf16_t* __restrict__ W1i,
    const bf16_t* __restrict__ W1h, const bf16_t* __restrict__ W2i,
    const bf16_t* __restrict__ W2h, const bf16_t* __restrict__ Wfc,
    float* __restrict__ mel, float* __restrict__ attnw) {
  __shared__ float sh_a[BT * DDIM];  // attention GRU hidden (f32)
  __shared__ float sh1[BT * DDIM];   // GRU1 hidden
  __shared__ float sh2[BT * DDIM];   // GRU2 hidden
  __shared__ float sq[BT * DDIM];    // attention query
  __shared__ float sw[BT * TENC];    // energies -> softmax weights
  __shared__ float sv[DDIM];         // attn_v
  __shared__ bf16_t sHb[BT * DDIM];  // bf16 A-operand staging (hiddens / out)
  __shared__ union Arena {
    bf16_t x2[BT * 512];  // [attn_h | ctx] for GRU1
    struct {
      bf16_t x[BT * 96];    // x_t zero-padded 80->96
      bf16_t p1[BT * 256];  // prenet layer1 (also h1 bf16 for GRU2)
      bf16_t p[BT * 128];   // prenet out
    } pre;
  } ar;

  const int tid = threadIdx.x;
  const int lane = tid & 31;
  const int wave = tid >> 5;
  const int b0 = blockIdx.x * BT;

  for (int i = tid; i < BT * DDIM; i += WG_THREADS) {
    sh_a[i] = 0.f; sh1[i] = 0.f; sh2[i] = 0.f;
  }
  if (tid < DDIM) sv[tid] = attn_v[tid];
  __syncthreads();

  for (int step = 0; step < STEPS; ++step) {
    // 1: teacher-forced input x_t (zeros at step 0), pad K to 96
    for (int i = tid; i < BT * 96; i += WG_THREADS) {
      int m = i / 96, k = i - m * 96;
      float v = 0.f;
      if (step > 0 && k < NMEL)
        v = dec[((size_t)(b0 + m) * 800 + 2 * step) * NMEL + k];
      ar.pre.x[m * 96 + k] = (bf16_t)v;
    }
    __syncthreads();
    // 2-3: prenet (relu fused into WMMA epilogue, bf16 out)
    gemm_relu_bf16(ar.pre.x, 96, 3, Wp1, 16, pre_b1, ar.pre.p1, 256, wave, lane);
    __syncthreads();
    gemm_relu_bf16(ar.pre.p1, 256, 8, Wp2, 8, pre_b2, ar.pre.p, 128, wave, lane);
    __syncthreads();
    // 4-5: attention GRU
    for (int i = tid; i < BT * DDIM; i += WG_THREADS) sHb[i] = (bf16_t)sh_a[i];
    __syncthreads();
    gru_fused(ar.pre.p, 128, 4, Wai, sHb, Wah, arnn_bih, arnn_bhh, sh_a, wave, lane);
    __syncthreads();
    // 6-7: q = attn_h @ Wq.T + bq
    for (int i = tid; i < BT * DDIM; i += WG_THREADS) sHb[i] = (bf16_t)sh_a[i];
    __syncthreads();
    gemm_bias_f32(sHb, 256, 8, Wq, 16, attn_bq, sq, 256, wave, lane);
    __syncthreads();
    // 8: Bahdanau attention, one wave per batch row
    {
      int bl = wave;
      int b = b0 + bl;
      const float* krow = keys + (size_t)b * TENC * DDIM;
      float qreg[8];
#pragma unroll
      for (int j = 0; j < 8; ++j) qreg[j] = sq[bl * DDIM + lane + 32 * j];
      for (int t = 0; t < TENC; ++t) {
        const float* kr = krow + (size_t)t * DDIM;
        float s = 0.f;
#pragma unroll
        for (int j = 0; j < 8; ++j) {
          int d = lane + 32 * j;
          s += tanhf(kr[d] + qreg[j]) * sv[d];
        }
        s = wred_add(s);
        if (lane == 0) sw[bl * TENC + t] = s;
      }
      // softmax over 256 (each lane owns 8 t-slots)
      float ev[8];
      float mx = -3.4e38f;
#pragma unroll
      for (int j = 0; j < 8; ++j) {
        ev[j] = sw[bl * TENC + lane + 32 * j];
        mx = fmaxf(mx, ev[j]);
      }
      mx = wred_max(mx);
      float sum = 0.f;
#pragma unroll
      for (int j = 0; j < 8; ++j) { ev[j] = __expf(ev[j] - mx); sum += ev[j]; }
      sum = wred_add(sum);
      float inv = 1.f / sum;
#pragma unroll
      for (int j = 0; j < 8; ++j) {
        int t = lane + 32 * j;
        float wv = ev[j] * inv;
        sw[bl * TENC + t] = wv;
        attnw[((size_t)b * TENC + t) * STEPS + step] = wv;  // (B,Tenc,steps)
      }
      // ctx = sum_t w[t]*enc[b,t,:]; build x2 = [attn_h | ctx] in bf16
      float cacc[8];
#pragma unroll
      for (int j = 0; j < 8; ++j) cacc[j] = 0.f;
      const float* erow = enc + (size_t)b * TENC * DDIM;
      for (int t = 0; t < TENC; ++t) {
        float wv = sw[bl * TENC + t];
        const float* er = erow + (size_t)t * DDIM;
#pragma unroll
        for (int j = 0; j < 8; ++j) cacc[j] += wv * er[lane + 32 * j];
      }
#pragma unroll
      for (int j = 0; j < 8; ++j) {
        int d = lane + 32 * j;
        ar.x2[bl * 512 + d] = (bf16_t)sh_a[bl * DDIM + d];
        ar.x2[bl * 512 + 256 + d] = (bf16_t)cacc[j];
      }
    }
    __syncthreads();
    // 9-10: GRU1 (K_x = 512)
    for (int i = tid; i < BT * DDIM; i += WG_THREADS) sHb[i] = (bf16_t)sh1[i];
    __syncthreads();
    gru_fused(ar.x2, 512, 16, W1i, sHb, W1h, g1_bih, g1_bhh, sh1, wave, lane);
    __syncthreads();
    // 11-12: GRU2 (x = new h1, h = h2)
    for (int i = tid; i < BT * DDIM; i += WG_THREADS) {
      ar.pre.p1[i] = (bf16_t)sh1[i];
      sHb[i] = (bf16_t)sh2[i];
    }
    __syncthreads();
    gru_fused(ar.pre.p1, 256, 8, W2i, sHb, W2h, g2_bih, g2_bhh, sh2, wave, lane);
    __syncthreads();
    // 13: out = h1 + h2 (bf16 A for fc)
    for (int i = tid; i < BT * DDIM; i += WG_THREADS)
      sHb[i] = (bf16_t)(sh1[i] + sh2[i]);
    __syncthreads();
    // 14: fc -> mel frames (y[b, m*2+r] -> mel[b, step*2+r, m])
    if (wave < 10) {
      int nt = wave;
      v8f acc = zero8();
      for (int kt = 0; kt < 8; ++kt) {
        v16bf a = load_frag_lds(sHb, 256, kt, lane);
        v16bf bmat = load_frag_w(Wfc, 8, nt, kt, lane);
        acc = wmma_bf16(a, bmat, acc);
      }
      int m0 = (lane >> 4) << 3;
      int col = nt * 16 + (lane & 15);
      float bv = fc_b[col];
      int mcol = col >> 1, rf = col & 1;
#pragma unroll
      for (int r = 0; r < 8; ++r) {
        int b = b0 + m0 + r;
        mel[((size_t)b * 800 + step * 2 + rf) * NMEL + mcol] = acc[r] + bv;
      }
    }
    __syncthreads();
  }
}

// ---------------------------------------------------------------------------
extern "C" void kernel_launch(void* const* d_in, const int* in_sizes, int n_in,
                              void* d_out, int out_size, void* d_ws,
                              size_t ws_size, hipStream_t stream) {
  (void)in_sizes; (void)n_in; (void)out_size; (void)ws_size;
  const float* enc      = (const float*)d_in[0];
  const float* dec      = (const float*)d_in[1];
  const float* pre_w1   = (const float*)d_in[2];
  const float* pre_b1   = (const float*)d_in[3];
  const float* pre_w2   = (const float*)d_in[4];
  const float* pre_b2   = (const float*)d_in[5];
  const float* attn_wq  = (const float*)d_in[6];
  const float* attn_bq  = (const float*)d_in[7];
  const float* attn_wk  = (const float*)d_in[8];
  const float* attn_bk  = (const float*)d_in[9];
  const float* attn_v   = (const float*)d_in[10];
  const float* arnn_wih = (const float*)d_in[11];
  const float* arnn_whh = (const float*)d_in[12];
  const float* arnn_bih = (const float*)d_in[13];
  const float* arnn_bhh = (const float*)d_in[14];
  const float* g1_wih   = (const float*)d_in[15];
  const float* g1_whh   = (const float*)d_in[16];
  const float* g1_bih   = (const float*)d_in[17];
  const float* g1_bhh   = (const float*)d_in[18];
  const float* g2_wih   = (const float*)d_in[19];
  const float* g2_whh   = (const float*)d_in[20];
  const float* g2_bih   = (const float*)d_in[21];
  const float* g2_bhh   = (const float*)d_in[22];
  const float* fc_w     = (const float*)d_in[23];
  const float* fc_b     = (const float*)d_in[24];

  // workspace: [keys f32: 64*256*256] [bf16 weight fragments]
  float* keysbuf = (float*)d_ws;
  bf16_t* fr = (bf16_t*)((char*)d_ws + (size_t)64 * 256 * 256 * sizeof(float));
  size_t off = 0;
  auto pack = [&](const float* src, int N, int K, int Ktiles) -> bf16_t* {
    bf16_t* dst = fr + off;
    int total = (N / 16) * Ktiles * 32;
    pack_weight<<<(total + 255) / 256, 256, 0, stream>>>(src, dst, N, K, Ktiles);
    off += (size_t)(N / 16) * Ktiles * 512;
    return dst;
  };
  bf16_t* Wp1 = pack(pre_w1, 256, 80, 3);    // K padded 80 -> 96
  bf16_t* Wp2 = pack(pre_w2, 128, 256, 8);
  bf16_t* Wai = pack(arnn_wih, 768, 128, 4);
  bf16_t* Wah = pack(arnn_whh, 768, 256, 8);
  bf16_t* Wq  = pack(attn_wq, 256, 256, 8);
  bf16_t* W1i = pack(g1_wih, 768, 512, 16);
  bf16_t* W1h = pack(g1_whh, 768, 256, 8);
  bf16_t* W2i = pack(g2_wih, 768, 256, 8);
  bf16_t* W2h = pack(g2_whh, 768, 256, 8);
  bf16_t* Wfc = pack(fc_w, 160, 256, 8);

  keys_kernel<<<64 * 256, 256, 0, stream>>>(enc, attn_wk, attn_bk, keysbuf);

  float* mel = (float*)d_out;                  // (64, 800, 80)
  float* attnw = mel + (size_t)64 * 800 * 80;  // (64, 256, 400)

  decoder_kernel<<<4, WG_THREADS, 0, stream>>>(
      enc, dec, keysbuf, pre_b1, pre_b2, attn_bq, attn_v, arnn_bih, arnn_bhh,
      g1_bih, g1_bhh, g2_bih, g2_bhh, fc_b, Wp1, Wp2, Wai, Wah, Wq, W1i, W1h,
      W2i, W2h, Wfc, mel, attnw);
}
